// IndicNationalSTGNN_26353919328980
// MI455X (gfx1250) — matrix operand order repaired and checked by reading, not compiled, not run on hardware
//
#include <hip/hip_runtime.h>
#include <hip/hip_bf16.h>

typedef __attribute__((ext_vector_type(16))) _Float16 v16h;
typedef __attribute__((ext_vector_type(8)))  float    v8f;

#define N_NODES 50000
#define N_EDGES 1600000
#define E_TOT   (N_EDGES + N_NODES)
#define BATCH   512
#define SEQ     64
#define FF      2048
#define MROWS   (BATCH * SEQ)

// ---------- ordered-uint encoding for float atomicMax ----------
__device__ __forceinline__ unsigned f2ord(float f) {
  unsigned b = __float_as_uint(f);
  return (b & 0x80000000u) ? ~b : (b | 0x80000000u);
}
__device__ __forceinline__ float ord2f(unsigned e) {
  unsigned b = (e & 0x80000000u) ? (e ^ 0x80000000u) : ~e;
  return __uint_as_float(b);
}

// ---------- init accumulators (ws is poisoned by the harness) ----------
__global__ void k_init(float* __restrict__ acc, float* __restrict__ den,
                       unsigned* __restrict__ m, float* __restrict__ gsum) {
  int i = blockIdx.x * blockDim.x + threadIdx.x;
  if (i < N_NODES * 32) acc[i] = 0.f;
  if (i < N_NODES * 2) { den[i] = 0.f; m[i] = 0x007FFFFFu; /* enc(-inf) */ }
  if (i < 32) gsum[i] = 0.f;
}

// ---------- GAT: per-node projection + attention logits ----------
__global__ void k_gat_node(const float* __restrict__ x, const float* __restrict__ w,
                           const float* __restrict__ asrc_w, const float* __restrict__ adst_w,
                           float* __restrict__ xp, float* __restrict__ a_src,
                           float* __restrict__ a_dst) {
  int n = blockIdx.x * blockDim.x + threadIdx.x;
  if (n >= N_NODES) return;
  float x0 = x[n * 3], x1 = x[n * 3 + 1], x2 = x[n * 3 + 2];
  float as[2] = {0.f, 0.f}, ad[2] = {0.f, 0.f};
#pragma unroll
  for (int c = 0; c < 32; ++c) {
    float v = fmaf(w[c * 3], x0, fmaf(w[c * 3 + 1], x1, w[c * 3 + 2] * x2));
    xp[n * 32 + c] = v;
    int h = c >> 4;                    // (2,16) flat == global channel c
    as[h] = fmaf(v, asrc_w[c], as[h]);
    ad[h] = fmaf(v, adst_w[c], ad[h]);
  }
  a_src[n * 2] = as[0]; a_src[n * 2 + 1] = as[1];
  a_dst[n * 2] = ad[0]; a_dst[n * 2 + 1] = ad[1];
}

__device__ __forceinline__ void edge_alpha(const int* __restrict__ ei,
                                           const float* __restrict__ a_src,
                                           const float* __restrict__ a_dst,
                                           int e, int& s, int& dn, float al[2]) {
  if (e < N_EDGES) { s = ei[e]; dn = ei[N_EDGES + e]; }
  else             { s = dn = e - N_EDGES; }          // self loop
#pragma unroll
  for (int h = 0; h < 2; ++h) {
    float a = a_src[s * 2 + h] + a_dst[dn * 2 + h];
    al[h] = a > 0.f ? a : 0.2f * a;                   // leaky_relu(0.2)
  }
}

__global__ void k_gat_max(const int* __restrict__ ei, const float* __restrict__ a_src,
                          const float* __restrict__ a_dst, unsigned* __restrict__ m) {
  int e = blockIdx.x * blockDim.x + threadIdx.x;
  if (e >= E_TOT) return;
  int s, dn; float al[2];
  edge_alpha(ei, a_src, a_dst, e, s, dn, al);
  atomicMax(&m[dn * 2 + 0], f2ord(al[0]));
  atomicMax(&m[dn * 2 + 1], f2ord(al[1]));
}

__global__ void k_gat_sum(const int* __restrict__ ei, const float* __restrict__ a_src,
                          const float* __restrict__ a_dst, const unsigned* __restrict__ m,
                          float* __restrict__ den) {
  int e = blockIdx.x * blockDim.x + threadIdx.x;
  if (e >= E_TOT) return;
  int s, dn; float al[2];
  edge_alpha(ei, a_src, a_dst, e, s, dn, al);
  atomicAdd(&den[dn * 2 + 0], __expf(al[0] - ord2f(m[dn * 2 + 0])));
  atomicAdd(&den[dn * 2 + 1], __expf(al[1] - ord2f(m[dn * 2 + 1])));
}

__global__ void k_gat_acc(const int* __restrict__ ei, const float* __restrict__ a_src,
                          const float* __restrict__ a_dst, const unsigned* __restrict__ m,
                          const float* __restrict__ den, const float* __restrict__ xp,
                          float* __restrict__ acc) {
  int e = blockIdx.x * blockDim.x + threadIdx.x;
  if (e >= E_TOT) return;
  int s, dn; float al[2];
  edge_alpha(ei, a_src, a_dst, e, s, dn, al);
  float c0 = __expf(al[0] - ord2f(m[dn * 2 + 0])) / den[dn * 2 + 0];
  float c1 = __expf(al[1] - ord2f(m[dn * 2 + 1])) / den[dn * 2 + 1];
  const float* xs = xp + s * 32;
  float* ac = acc + dn * 32;
#pragma unroll
  for (int c = 0; c < 16; ++c)  atomicAdd(&ac[c],      xs[c]      * c0);
#pragma unroll
  for (int c = 0; c < 16; ++c)  atomicAdd(&ac[16 + c], xs[16 + c] * c1);
}

// relu(acc + bias) then mean over nodes (sum here, divide in final kernel)
__global__ void k_gat_reduce(const float* __restrict__ acc, const float* __restrict__ bias,
                             float* __restrict__ gsum) {
  __shared__ float sd[128 * 32];
  int tid = threadIdx.x;
  float loc[32];
#pragma unroll
  for (int c = 0; c < 32; ++c) loc[c] = 0.f;
  for (int n = blockIdx.x * blockDim.x + tid; n < N_NODES; n += gridDim.x * blockDim.x) {
#pragma unroll
    for (int c = 0; c < 32; ++c) loc[c] += fmaxf(acc[n * 32 + c] + bias[c], 0.f);
  }
#pragma unroll
  for (int c = 0; c < 32; ++c) sd[tid * 32 + c] = loc[c];
  __syncthreads();
  for (int st = 64; st > 0; st >>= 1) {
    if (tid < st)
      for (int c = 0; c < 32; ++c) sd[tid * 32 + c] += sd[(tid + st) * 32 + c];
    __syncthreads();
  }
  if (tid < 32) atomicAdd(&gsum[tid], sd[tid]);
}

// ---------- env encoder: fused MHA (nhead=2, Dh=2) + residual + LN1 ----------
__global__ void k_env_attn(const float* __restrict__ xin, float* __restrict__ xout,
                           const float* __restrict__ inw, const float* __restrict__ inb,
                           const float* __restrict__ ow,  const float* __restrict__ ob,
                           const float* __restrict__ n1w, const float* __restrict__ n1b) {
  __shared__ float sK[SEQ * 4], sV[SEQ * 4];
  int b = blockIdx.x, t = threadIdx.x;
  const float* xr = xin + (b * SEQ + t) * 4;
  float x4[4] = {xr[0], xr[1], xr[2], xr[3]};
  float qkv[12];
#pragma unroll
  for (int j = 0; j < 12; ++j)
    qkv[j] = fmaf(inw[j * 4 + 3], x4[3], fmaf(inw[j * 4 + 2], x4[2],
             fmaf(inw[j * 4 + 1], x4[1], fmaf(inw[j * 4], x4[0], inb[j]))));
#pragma unroll
  for (int d = 0; d < 4; ++d) { sK[t * 4 + d] = qkv[4 + d]; sV[t * 4 + d] = qkv[8 + d]; }
  __syncthreads();
  const float scale = 0.7071067811865475f;   // 1/sqrt(Dh=2)
  float o[4];
#pragma unroll
  for (int h = 0; h < 2; ++h) {
    float q0 = qkv[h * 2], q1 = qkv[h * 2 + 1];
    float mx = -3.0e38f;
    for (int k = 0; k < SEQ; ++k) {
      float sc = (q0 * sK[k * 4 + h * 2] + q1 * sK[k * 4 + h * 2 + 1]) * scale;
      mx = fmaxf(mx, sc);
    }
    float sum = 0.f, o0 = 0.f, o1 = 0.f;
    for (int k = 0; k < SEQ; ++k) {
      float sc = (q0 * sK[k * 4 + h * 2] + q1 * sK[k * 4 + h * 2 + 1]) * scale;
      float ee = __expf(sc - mx);
      sum += ee;
      o0 = fmaf(ee, sV[k * 4 + h * 2], o0);
      o1 = fmaf(ee, sV[k * 4 + h * 2 + 1], o1);
    }
    o[h * 2] = o0 / sum; o[h * 2 + 1] = o1 / sum;
  }
  float y[4], mean = 0.f;
#pragma unroll
  for (int d = 0; d < 4; ++d) {
    float a = fmaf(ow[d * 4 + 3], o[3], fmaf(ow[d * 4 + 2], o[2],
              fmaf(ow[d * 4 + 1], o[1], fmaf(ow[d * 4], o[0], ob[d]))));
    y[d] = x4[d] + a;
    mean += y[d];
  }
  mean *= 0.25f;
  float var = 0.f;
#pragma unroll
  for (int d = 0; d < 4; ++d) { float c = y[d] - mean; var = fmaf(c, c, var); }
  var *= 0.25f;
  float r = rsqrtf(var + 1e-5f);
  float* op = xout + (b * SEQ + t) * 4;
#pragma unroll
  for (int d = 0; d < 4; ++d) op[d] = fmaf((y[d] - mean) * r, n1w[d], n1b[d]);
}

// ---------- pre-convert l2_w to zero-padded f16 B image [2][16][2048] ----------
__global__ void k_prep_w2(const float* __restrict__ l2w, _Float16* __restrict__ w2h) {
  int i = blockIdx.x * blockDim.x + threadIdx.x;
  if (i >= 2 * 16 * FF) return;
  int L = i / (16 * FF);
  int n = (i / FF) & 15;
  int k = i % FF;
  float v = (n < 4) ? l2w[(L * 4 + n) * FF + k] : 0.f;
  w2h[i] = (_Float16)v;
}

// ---------- fused FFN1(relu) -> FFN2 (WMMA f16, f32 accum) + residual + LN2 ----
__global__ void __launch_bounds__(128) k_env_ffn(
    const float* __restrict__ xin, float* __restrict__ xout,
    const float* __restrict__ W1, const float* __restrict__ b1,
    const _Float16* __restrict__ W2h, const float* __restrict__ b2,
    const float* __restrict__ n2w, const float* __restrict__ n2b) {
  __shared__ float sW1[FF * 4];
  __shared__ float sB1[FF];
  __shared__ float sC[4 * 16 * 4];
  int tid = threadIdx.x;
  // Cooperatively stage W1 (32KB) + b1 (8KB) in LDS (reused 64 rows / block).
  for (int i = tid; i < FF; i += 128) {
    ((float4*)sW1)[i] = ((const float4*)W1)[i];
    sB1[i] = b1[i];
  }
  __syncthreads();

  int wave = tid >> 5, lane = tid & 31;
  int hi = lane >> 4, mrow = lane & 15;
  int tile = blockIdx.x * 4 + wave;       // 16 rows per wave
  int row = tile * 16 + mrow;
  const float4 xv = *(const float4*)(xin + row * 4);
  float x0 = xv.x, x1 = xv.y, x2 = xv.z, x3 = xv.w;

  int nB = mrow;                          // B-fragment column for this lane
  const _Float16* w2row = W2h + nB * FF + hi * 16;

  v8f acc = {};
#pragma unroll 1
  for (int chunk = 0; chunk < FF / 32; ++chunk) {
    int k0 = chunk * 32;
    // Build A-fragment: f1[m, k0+kloc] = relu(x . W1[k] + b1[k])  (contiguous K: lanes 0-15 -> K 0-15, 16-31 -> K 16-31)
    v16h a;
#pragma unroll
    for (int v = 0; v < 16; ++v) {
      int k = k0 + hi * 16 + v;
      const float* wv = sW1 + k * 4;
      float hval = fmaf(x3, wv[3], fmaf(x2, wv[2], fmaf(x1, wv[1], fmaf(x0, wv[0], sB1[k]))));
      a[v] = (_Float16)fmaxf(hval, 0.f);
    }
    // B-fragment: 16 contiguous halves, lanes 0-15 -> K 0-15, 16-31 -> K 16-31
    v16h bfrag = *(const v16h*)(w2row + k0);
    acc = __builtin_amdgcn_wmma_f32_16x16x32_f16(false, a, false, bfrag,
                                                 (short)0, acc, false, false);
  }
  // Spill C tile (only n<4 valid) for per-row LayerNorm.
  if (nB < 4) {
#pragma unroll
    for (int j = 0; j < 8; ++j) sC[(wave * 16 + (j + 8 * hi)) * 4 + nB] = acc[j];
  }
  __syncthreads();
  if (lane < 16) {
    int r2 = tile * 16 + lane;
    float y[4], mean = 0.f;
#pragma unroll
    for (int d = 0; d < 4; ++d) {
      y[d] = xin[r2 * 4 + d] + sC[(wave * 16 + lane) * 4 + d] + b2[d];
      mean += y[d];
    }
    mean *= 0.25f;
    float var = 0.f;
#pragma unroll
    for (int d = 0; d < 4; ++d) { float c = y[d] - mean; var = fmaf(c, c, var); }
    var *= 0.25f;
    float rr = rsqrtf(var + 1e-5f);
#pragma unroll
    for (int d = 0; d < 4; ++d) xout[r2 * 4 + d] = fmaf((y[d] - mean) * rr, n2w[d], n2b[d]);
  }
}

// ---------- final: forecast (constant, from collapsed ts encoder) + sigmoid head
__global__ void k_final(const float* __restrict__ tbuf, const float* __restrict__ gsum,
                        const float* __restrict__ dec_w, const float* __restrict__ dec_b,
                        const float* __restrict__ ts_n2_b, const float* __restrict__ fc_w,
                        const float* __restrict__ fc_b, float* __restrict__ out) {
  int b = blockIdx.x * blockDim.x + threadIdx.x;
  if (b >= BATCH) return;
  float tsb = ts_n2_b[1];                 // LN over dim-1 outputs its bias exactly
  float fore[7];
#pragma unroll
  for (int j = 0; j < 7; ++j) {
    fore[j] = fmaf(tsb, dec_w[j], dec_b[j]);
    out[BATCH + b * 7 + j] = fore[j];
  }
  float z = fc_b[0];
  const float invN = 1.0f / (float)N_NODES;
  for (int c = 0; c < 32; ++c) z = fmaf(fc_w[c], gsum[c] * invN, z);
  const float* t = tbuf + (b * SEQ + (SEQ - 1)) * 4;
#pragma unroll
  for (int d = 0; d < 4; ++d) z = fmaf(fc_w[32 + d], t[d], z);
#pragma unroll
  for (int j = 0; j < 7; ++j) z = fmaf(fc_w[36 + j], fore[j], z);
  out[b] = 1.0f / (1.0f + __expf(-z));
}

extern "C" void kernel_launch(void* const* d_in, const int* in_sizes, int n_in,
                              void* d_out, int out_size, void* d_ws, size_t ws_size,
                              hipStream_t stream) {
  (void)in_sizes; (void)n_in; (void)out_size; (void)ws_size;
  const float* x        = (const float*)d_in[0];
  const int*   ei       = (const int*)  d_in[1];
  const float* e        = (const float*)d_in[2];
  const float* ts_n2_b  = (const float*)d_in[15];
  const float* env_in_w = (const float*)d_in[16];
  const float* env_in_b = (const float*)d_in[17];
  const float* env_ow   = (const float*)d_in[18];
  const float* env_ob   = (const float*)d_in[19];
  const float* env_l1w  = (const float*)d_in[20];
  const float* env_l1b  = (const float*)d_in[21];
  const float* env_l2w  = (const float*)d_in[22];
  const float* env_l2b  = (const float*)d_in[23];
  const float* env_n1w  = (const float*)d_in[24];
  const float* env_n1b  = (const float*)d_in[25];
  const float* env_n2w  = (const float*)d_in[26];
  const float* env_n2b  = (const float*)d_in[27];
  const float* dec_w    = (const float*)d_in[28];
  const float* dec_b    = (const float*)d_in[29];
  const float* gat_w    = (const float*)d_in[30];
  const float* att_src  = (const float*)d_in[31];
  const float* att_dst  = (const float*)d_in[32];
  const float* gat_b    = (const float*)d_in[33];
  const float* fc_w     = (const float*)d_in[34];
  const float* fc_b     = (const float*)d_in[35];
  float* out = (float*)d_out;

  char* wsb = (char*)d_ws;
  size_t off = 0;
  auto carve = [&](size_t bytes) -> char* {
    char* p = wsb + off;
    off = (off + bytes + 255) & ~(size_t)255;
    return p;
  };
  float*    xp    = (float*)   carve(N_NODES * 32 * sizeof(float));
  float*    a_src = (float*)   carve(N_NODES * 2 * sizeof(float));
  float*    a_dst = (float*)   carve(N_NODES * 2 * sizeof(float));
  unsigned* mEnc  = (unsigned*)carve(N_NODES * 2 * sizeof(unsigned));
  float*    den   = (float*)   carve(N_NODES * 2 * sizeof(float));
  float*    accb  = (float*)   carve(N_NODES * 32 * sizeof(float));
  float*    gsum  = (float*)   carve(32 * sizeof(float));
  float*    bufA  = (float*)   carve(MROWS * 4 * sizeof(float));
  float*    bufB  = (float*)   carve(MROWS * 4 * sizeof(float));
  _Float16* w2h   = (_Float16*)carve(2 * 16 * FF * sizeof(_Float16));

  // weight prep + accumulator init
  k_prep_w2<<<(2 * 16 * FF + 255) / 256, 256, 0, stream>>>(env_l2w, w2h);
  k_init<<<(N_NODES * 32 + 255) / 256, 256, 0, stream>>>(accb, den, mEnc, gsum);

  // GAT branch
  k_gat_node<<<(N_NODES + 255) / 256, 256, 0, stream>>>(x, gat_w, att_src, att_dst,
                                                        xp, a_src, a_dst);
  int eb = (E_TOT + 255) / 256;
  k_gat_max<<<eb, 256, 0, stream>>>(ei, a_src, a_dst, mEnc);
  k_gat_sum<<<eb, 256, 0, stream>>>(ei, a_src, a_dst, mEnc, den);
  k_gat_acc<<<eb, 256, 0, stream>>>(ei, a_src, a_dst, mEnc, den, xp, accb);
  k_gat_reduce<<<128, 128, 0, stream>>>(accb, gat_b, gsum);

  // env transformer encoder, 2 layers (attn+LN1, then WMMA-fused FFN+LN2)
  k_env_attn<<<BATCH, SEQ, 0, stream>>>(e, bufA, env_in_w, env_in_b, env_ow, env_ob,
                                        env_n1w, env_n1b);
  k_env_ffn<<<MROWS / 64, 128, 0, stream>>>(bufA, bufB, env_l1w, env_l1b, w2h,
                                            env_l2b, env_n2w, env_n2b);
  k_env_attn<<<BATCH, SEQ, 0, stream>>>(bufB, bufA, env_in_w + 48, env_in_b + 12,
                                        env_ow + 16, env_ob + 4, env_n1w + 4, env_n1b + 4);
  k_env_ffn<<<MROWS / 64, 128, 0, stream>>>(bufA, bufB, env_l1w + FF * 4, env_l1b + FF,
                                            w2h + 16 * FF, env_l2b + 4, env_n2w + 4,
                                            env_n2b + 4);

  // output head (forecast is the collapsed ts-encoder constant)
  k_final<<<(BATCH + 255) / 256, 256, 0, stream>>>(bufB, gsum, dec_w, dec_b, ts_n2_b,
                                                   fc_w, fc_b, out);
}
